// HRTBUnit_63453846831811
// MI455X (gfx1250) — matrix operand order repaired and checked by reading, not compiled, not run on hardware
//
#include <hip/hip_runtime.h>
#include <hip/hip_bf16.h>

// ---------------------------------------------------------------------------
// HRTB unit for MI455X (gfx1250, wave32).  All dense linear algebra goes
// through V_WMMA_F32_16X16X32_F16 (f16 operands, f32 accum).  Spatial
// attention is flash-fused (4096x4096 score matrix never materialized:
// it would be 268MB > 192MB L2).  Convs are implicit-GEMM with compile-time
// kernel size (no runtime divides in the hot loop) and 32x16 register tiles
// (one weight B-fragment feeds two WMMAs).  Depthwise convs / softmax / LN /
// gating are VALU kernels.  Workspace requirement ~100 MB.
// ---------------------------------------------------------------------------

typedef __attribute__((ext_vector_type(16))) _Float16 v16h;
typedef __attribute__((ext_vector_type(8)))  float    v8f;
typedef __attribute__((ext_vector_type(2)))  _Float16 h2t;

#define NB   4
#define CD   64
#define HWSZ 4096
#define HIDC 192

union HU { v16h v; h2t h[8]; };

__device__ __forceinline__ float gelu_f(float x){
  return 0.5f * x * (1.0f + erff(x * 0.70710678118654752f));
}
__device__ __forceinline__ float sigm_f(float x){
  return 1.0f / (1.0f + __expf(-x));
}

__device__ __forceinline__ v8f wmma_f16(v16h a, v16h b, v8f c){
  // (neg_a, A, neg_b, B, c_mod, C, reuse_a, reuse_b)
  return __builtin_amdgcn_wmma_f32_16x16x32_f16(false, a, false, b, (short)0, c, false, false);
}

// A fragment: 16x32 f16.  Element (m,k) at src[m*ld + k] (src pre-offset to
// (m0,k0)).  Lane L (0..15): M=L, K in {0..7,16..23}; lane L+16: K {8..15,24..31}.
__device__ __forceinline__ v16h loadA16(const _Float16* src, int ld){
  int lane = threadIdx.x & 31;
  const _Float16* p = src + (size_t)(lane & 15) * ld + (lane >> 4) * 8;
  HU u;
#pragma unroll
  for (int v = 0; v < 8; ++v){
    int kk = (v < 4) ? (2 * v) : (16 + 2 * (v - 4));
    u.h[v] = *(const h2t*)(p + kk);
  }
  return u.v;
}

// B fragment: 32x16 f16 with storage [col][k]: element (k,n) at src[n*ld + k].
// Lane n (0..15): N=n, K in {0..7,16..23}; lane n+16: K {8..15,24..31}.
__device__ __forceinline__ v16h loadB16(const _Float16* src, int ld){
  int lane = threadIdx.x & 31;
  const _Float16* p = src + (size_t)(lane & 15) * ld + (lane >> 4) * 8;
  HU u;
#pragma unroll
  for (int v = 0; v < 8; ++v)
    u.h[v] = *(const h2t*)(p + ((v < 4) ? (2 * v) : (16 + 2 * (v - 4))));
  return u.v;
}

// Implicit-im2col A fragment for KSxKS conv over [HW][IC] f16 activations,
// K-order = (ky,kx,ic) so f16 pairs stay contiguous in ic.  KS / icShift are
// compile-time: the kpos/KS division strength-reduces to mul+shift.
template<int KS, int ICS>
__device__ __forceinline__ v16h loadA_im2col(const _Float16* inb, int pix0,
                                             int pad, int dil, int k0){
  int lane = threadIdx.x & 31;
  int pix  = pix0 + (lane & 15);
  int y = pix >> 6, x = pix & 63;
  int hi  = lane >> 4;
  const int icm = (1 << ICS) - 1;
  HU u;
#pragma unroll
  for (int v = 0; v < 8; ++v){
    int kk   = k0 + hi * 8 + ((v < 4) ? (2 * v) : (16 + 2 * (v - 4)));
    int kpos = kk >> ICS;
    int ic   = kk & icm;
    int ky   = kpos / KS;
    int kx   = kpos - ky * KS;
    int sy = y + ky * dil - pad;
    int sx = x + kx * dil - pad;
    h2t val; val[0] = (_Float16)0.f; val[1] = (_Float16)0.f;
    if ((unsigned)sy < 64u && (unsigned)sx < 64u)
      val = *(const h2t*)(inb + (((size_t)((sy << 6) + sx)) << ICS) + ic);
    u.h[v] = val;
  }
  return u.v;
}

// ---------------------------------------------------------------------------
// 0) weight reorder  [OC][IC][KS][KS] f32  ->  [OC][KS*KS][IC] f16
// ---------------------------------------------------------------------------
__global__ void k_reorder_w(const float* w, _Float16* wr, int OC, int IC, int KS){
  int idx = blockIdx.x * 256 + threadIdx.x;
  int kk2 = KS * KS;
  int tot = OC * IC * kk2;
  if (idx >= tot) return;
  int oc  = idx / (IC * kk2);
  int rem = idx - oc * IC * kk2;
  int ic  = rem / kk2;
  int kp  = rem - ic * kk2;
  wr[(size_t)oc * IC * kk2 + (size_t)kp * IC + ic] = (_Float16)w[idx];
}

// ---------------------------------------------------------------------------
// 1) six depthwise 3x3 convs on x, emitting f16 in WMMA-friendly layouts:
//    Qc,Kc,vC: [b][c][n]   VcT,qT,kT: [b][n][c]
// ---------------------------------------------------------------------------
__global__ void k_dw6(const float* X,
                      const float* qw, const float* qb, const float* kw, const float* kb,
                      const float* vw, const float* vb,
                      const float* hqw, const float* hqb, const float* hkw, const float* hkb,
                      const float* hvw, const float* hvb,
                      _Float16* Qc, _Float16* Kc, _Float16* VcT,
                      _Float16* qT, _Float16* kT, _Float16* vC){
  int i = blockIdx.x * 256 + threadIdx.x;         // B*C*HW
  int n = i & 4095; int c = (i >> 12) & 63; int b = i >> 18;
  int y = n >> 6, x = n & 63;
  float s0=0,s1=0,s2=0,s3=0,s4=0,s5=0;
#pragma unroll
  for (int ky = 0; ky < 3; ++ky){
    int sy = y + ky - 1; if ((unsigned)sy >= 64u) continue;
#pragma unroll
    for (int kx = 0; kx < 3; ++kx){
      int sx = x + kx - 1; if ((unsigned)sx >= 64u) continue;
      float xv = X[(((size_t)b * 64 + c) << 12) + (sy << 6) + sx];
      int wi = c * 9 + ky * 3 + kx;
      s0 += xv * qw[wi];  s1 += xv * kw[wi];  s2 += xv * vw[wi];
      s3 += xv * hqw[wi]; s4 += xv * hkw[wi]; s5 += xv * hvw[wi];
    }
  }
  s0 += qb[c]; s1 += kb[c]; s2 += vb[c]; s3 += hqb[c]; s4 += hkb[c]; s5 += hvb[c];
  size_t cn = (((size_t)b * 64 + c) << 12) + n;
  size_t nc = ((((size_t)b << 12) + n) << 6) + c;
  Qc[cn] = (_Float16)s0;  Kc[cn] = (_Float16)s1;  VcT[nc] = (_Float16)s2;
  qT[nc] = (_Float16)s3;  kT[nc] = (_Float16)s4;  vC[cn] = (_Float16)s5;
}

// ---------------------------------------------------------------------------
// 2) channel attention:  S = scale * Q K^T  (M=N=64, K=4096)
// ---------------------------------------------------------------------------
__global__ void k_chan_qk(const _Float16* Qc, const _Float16* Kc, float* Sc, float scale){
  int b  = blockIdx.y;
  int c0 = (blockIdx.x >> 2) * 16;
  int d0 = (blockIdx.x & 3) * 16;
  const _Float16* Qb = Qc + ((size_t)b << 18);
  const _Float16* Kb = Kc + ((size_t)b << 18);
  v8f acc = {};
  for (int k0 = 0; k0 < 4096; k0 += 32){
    v16h a  = loadA16(Qb + (size_t)c0 * 4096 + k0, 4096);
    v16h bb = loadB16(Kb + (size_t)d0 * 4096 + k0, 4096);
    acc = wmma_f16(a, bb, acc);
  }
  int lane = threadIdx.x & 31; int nn = lane & 15; int hi = lane >> 4;
#pragma unroll
  for (int r = 0; r < 8; ++r)
    Sc[((size_t)b * 64 + c0 + r + hi * 8) * 64 + d0 + nn] = acc[r] * scale;
}

__global__ void k_chan_softmax(const float* Sc, _Float16* attn16){
  int t = threadIdx.x;                       // 256 = B*C rows
  const float* row = Sc + (size_t)t * 64;
  float mx = -3e38f;
  for (int d = 0; d < 64; ++d) mx = fmaxf(mx, row[d]);
  float sum = 0.f;
  for (int d = 0; d < 64; ++d) sum += __expf(row[d] - mx);
  float inv = 1.f / sum;
  _Float16* orow = attn16 + (size_t)t * 64;
  for (int d = 0; d < 64; ++d) orow[d] = (_Float16)(__expf(row[d] - mx) * inv);
}

// Fc^T[n][c] = sum_d VcT[n][d] * attn[c][d]   (M=n, N=c, K=64)
__global__ void k_chan_av(const _Float16* VcT, const _Float16* attn16, float* FcT){
  int b = blockIdx.z, ct = blockIdx.y, nt = blockIdx.x;
  const _Float16* Vb = VcT + ((size_t)b << 18);
  const _Float16* Ab = attn16 + ((size_t)b << 12);
  int n0 = nt * 16, c0 = ct * 16;
  v8f acc = {};
#pragma unroll
  for (int k0 = 0; k0 < 64; k0 += 32){
    v16h a  = loadA16(Vb + (size_t)n0 * 64 + k0, 64);
    v16h bb = loadB16(Ab + (size_t)c0 * 64 + k0, 64);
    acc = wmma_f16(a, bb, acc);
  }
  int lane = threadIdx.x & 31; int nn = lane & 15; int hi = lane >> 4;
#pragma unroll
  for (int r = 0; r < 8; ++r)
    FcT[((((size_t)b << 12) + n0 + r + hi * 8) << 6) + c0 + nn] = acc[r];
}

// ---------------------------------------------------------------------------
// 3) spatial attention, flash-style.  One wave owns 16 query rows, streams
//    all 4096 keys in 32-wide tiles; online softmax; P goes through LDS to
//    become an A-fragment for the PV WMMA.  8 WMMA per key tile.
// ---------------------------------------------------------------------------
__global__ void k_spatial_attn(const _Float16* qT, const _Float16* kT, const _Float16* vC,
                               float* aOutT, _Float16* aT16, float scale){
  __shared__ __align__(16) _Float16 Pl[16 * 32];
  int b  = blockIdx.y;
  int n0 = blockIdx.x * 16;
  const _Float16* qb = qT + ((size_t)b << 18);
  const _Float16* kb = kT + ((size_t)b << 18);
  const _Float16* vb = vC + ((size_t)b << 18);
  int lane = threadIdx.x & 31; int nn = lane & 15; int hi = lane >> 4;

  v16h a0 = loadA16(qb + (size_t)n0 * 64, 64);       // Q rows, K(c)=0..31
  v16h a1 = loadA16(qb + (size_t)n0 * 64 + 32, 64);  // K(c)=32..63
  v8f o0 = {}, o1 = {}, o2 = {}, o3 = {};
  float rmax[8], rsum[8];
#pragma unroll
  for (int r = 0; r < 8; ++r){ rmax[r] = -3e38f; rsum[r] = 0.f; }

  for (int m0 = 0; m0 < 4096; m0 += 32){
    v16h b00 = loadB16(kb + (size_t)m0 * 64, 64);
    v16h b01 = loadB16(kb + (size_t)m0 * 64 + 32, 64);
    v16h b10 = loadB16(kb + (size_t)(m0 + 16) * 64, 64);
    v16h b11 = loadB16(kb + (size_t)(m0 + 16) * 64 + 32, 64);
    v8f s0 = {}, s1 = {};
    s0 = wmma_f16(a0, b00, s0); s0 = wmma_f16(a1, b01, s0);
    s1 = wmma_f16(a0, b10, s1); s1 = wmma_f16(a1, b11, s1);

#pragma unroll
    for (int r = 0; r < 8; ++r){
      float x0 = s0[r] * scale, x1 = s1[r] * scale;
      float tm = fmaxf(x0, x1);
      tm = fmaxf(tm, __shfl_xor(tm, 1, 32));
      tm = fmaxf(tm, __shfl_xor(tm, 2, 32));
      tm = fmaxf(tm, __shfl_xor(tm, 4, 32));
      tm = fmaxf(tm, __shfl_xor(tm, 8, 32));
      float nm = fmaxf(rmax[r], tm);
      float al = __expf(rmax[r] - nm);
      float p0 = __expf(x0 - nm), p1 = __expf(x1 - nm);
      float ps = p0 + p1;
      ps += __shfl_xor(ps, 1, 32);
      ps += __shfl_xor(ps, 2, 32);
      ps += __shfl_xor(ps, 4, 32);
      ps += __shfl_xor(ps, 8, 32);
      rsum[r] = rsum[r] * al + ps;
      rmax[r] = nm;
      o0[r] *= al; o1[r] *= al; o2[r] *= al; o3[r] *= al;
      int row = r + hi * 8;
      Pl[row * 32 + nn]      = (_Float16)p0;   // C-layout -> A-layout via LDS
      Pl[row * 32 + 16 + nn] = (_Float16)p1;
    }
    __syncthreads();
    v16h pa = loadA16(Pl, 32);
    o0 = wmma_f16(pa, loadB16(vb + (size_t) 0 * 4096 + m0, 4096), o0);
    o1 = wmma_f16(pa, loadB16(vb + (size_t)16 * 4096 + m0, 4096), o1);
    o2 = wmma_f16(pa, loadB16(vb + (size_t)32 * 4096 + m0, 4096), o2);
    o3 = wmma_f16(pa, loadB16(vb + (size_t)48 * 4096 + m0, 4096), o3);
    __syncthreads();
  }
#pragma unroll
  for (int r = 0; r < 8; ++r){
    float inv = 1.f / rsum[r];
    size_t base = ((((size_t)b << 12) + n0 + r + hi * 8) << 6) + nn;
    float v0 = o0[r] * inv, v1 = o1[r] * inv, v2 = o2[r] * inv, v3 = o3[r] * inv;
    aOutT[base +  0] = v0; aT16[base +  0] = (_Float16)v0;
    aOutT[base + 16] = v1; aT16[base + 16] = (_Float16)v1;
    aOutT[base + 32] = v2; aT16[base + 32] = (_Float16)v2;
    aOutT[base + 48] = v3; aT16[base + 48] = (_Float16)v3;
  }
}

// ---------------------------------------------------------------------------
// 4) WMMA convs, implicit GEMM, 32x16 register tile (two pixel sub-tiles
//    share one weight B-fragment -> 2 WMMA per fragment fetch).
//    act: 0=none, 1=sigmoid, 2=gelu.
// ---------------------------------------------------------------------------
__device__ __forceinline__ void conv_epilogue(v8f acc0, v8f acc1, const float* bias,
                                              float* outF, int ldF, int offF,
                                              _Float16* outH, int ldH, int offH,
                                              int b, int pix0, int oc0, int act){
  int lane = threadIdx.x & 31; int nn = lane & 15; int hi = lane >> 4;
  int oc = oc0 + nn;
  float bv = bias[oc];
#pragma unroll
  for (int r = 0; r < 8; ++r){
    float v0 = acc0[r] + bv;
    float v1 = acc1[r] + bv;
    if (act == 1){ v0 = sigm_f(v0); v1 = sigm_f(v1); }
    else if (act == 2){ v0 = gelu_f(v0); v1 = gelu_f(v1); }
    size_t p0 = ((size_t)b << 12) + pix0 + r + hi * 8;
    size_t p1 = p0 + 16;
    if (outF){ outF[p0 * ldF + offF + oc] = v0; outF[p1 * ldF + offF + oc] = v1; }
    if (outH){ outH[p0 * ldH + offH + oc] = (_Float16)v0; outH[p1 * ldH + offH + oc] = (_Float16)v1; }
  }
}

// 1x1 conv (pure GEMM), runtime IC
__global__ void k_conv1x1(const _Float16* inT, const _Float16* wr, const float* bias,
                          float* outF, int ldF, int offF,
                          _Float16* outH, int ldH, int offH,
                          int IC, int act){
  int b    = blockIdx.z;
  int oc0  = blockIdx.y * 16;
  int pix0 = blockIdx.x * 32;
  const _Float16* inb = inT + (size_t)b * 4096 * IC;
  const _Float16* wp  = wr + (size_t)oc0 * IC;
  v8f acc0 = {}, acc1 = {};
  for (int k0 = 0; k0 < IC; k0 += 32){
    __builtin_prefetch(wp + k0 + 512, 0, 1);            // global_prefetch_b8
    v16h a0 = loadA16(inb + (size_t)pix0 * IC + k0, IC);
    v16h a1 = loadA16(inb + (size_t)(pix0 + 16) * IC + k0, IC);
    v16h bb = loadB16(wp + k0, IC);
    acc0 = wmma_f16(a0, bb, acc0);
    acc1 = wmma_f16(a1, bb, acc1);
  }
  conv_epilogue(acc0, acc1, bias, outF, ldF, offF, outH, ldH, offH, b, pix0, oc0, act);
}

// KSxKS conv, compile-time KS / icShift
template<int KS, int ICS>
__global__ void k_convK(const _Float16* inT, const _Float16* wr, const float* bias,
                        float* outF, int ldF, int offF,
                        _Float16* outH, int ldH, int offH,
                        int pad, int dil, int act){
  const int IC   = 1 << ICS;
  const int Ktot = IC * KS * KS;
  int b    = blockIdx.z;
  int oc0  = blockIdx.y * 16;
  int pix0 = blockIdx.x * 32;
  const _Float16* inb = inT + ((size_t)b * 4096 << ICS);
  const _Float16* wp  = wr + (size_t)oc0 * Ktot;
  v8f acc0 = {}, acc1 = {};
  for (int k0 = 0; k0 < Ktot; k0 += 32){
    __builtin_prefetch(wp + k0 + 512, 0, 1);            // global_prefetch_b8
    v16h a0 = loadA_im2col<KS, ICS>(inb, pix0,      pad, dil, k0);
    v16h a1 = loadA_im2col<KS, ICS>(inb, pix0 + 16, pad, dil, k0);
    v16h bb = loadB16(wp + k0, Ktot);
    acc0 = wmma_f16(a0, bb, acc0);
    acc1 = wmma_f16(a1, bb, acc1);
  }
  conv_epilogue(acc0, acc1, bias, outF, ldF, offF, outH, ldH, offH, b, pix0, oc0, act);
}

// ---------------------------------------------------------------------------
// 5) small VALU kernels: cwr depthwise, pool, gate, gating/eltwise, LN,
//    mixed depthwise refinement, residual.
// ---------------------------------------------------------------------------
__global__ void k_dw_nc(const float* in, const float* w, const float* bias, float* out){
  int i = blockIdx.x * 256 + threadIdx.x;      // [b][n][c]
  int c = i & 63; int n = (i >> 6) & 4095; int b = i >> 18;
  int y = n >> 6, x = n & 63;
  float s = bias[c];
#pragma unroll
  for (int ky = 0; ky < 3; ++ky){
    int sy = y + ky - 1; if ((unsigned)sy >= 64u) continue;
#pragma unroll
    for (int kx = 0; kx < 3; ++kx){
      int sx = x + kx - 1; if ((unsigned)sx >= 64u) continue;
      s += in[((((size_t)b << 12) + (sy << 6) + sx) << 6) + c] * w[c * 9 + ky * 3 + kx];
    }
  }
  out[i] = s;
}

__global__ void k_pool(const float* Fc2T, float* pooled){
  int bc = blockIdx.x;                          // B*64
  int b = bc >> 6, c = bc & 63;
  __shared__ float sh[256];
  float s = 0.f;
  for (int n = threadIdx.x; n < 4096; n += 256)
    s += Fc2T[((((size_t)b << 12) + n) << 6) + c];
  sh[threadIdx.x] = s; __syncthreads();
  for (int st = 128; st > 0; st >>= 1){
    if ((int)threadIdx.x < st) sh[threadIdx.x] += sh[threadIdx.x + st];
    __syncthreads();
  }
  if (threadIdx.x == 0) pooled[bc] = sh[0] * (1.f / 4096.f);
}

__global__ void k_gate(const float* pooled, const float* gw, const float* gb, float* gate){
  int t = threadIdx.x;                          // 256
  int b = t >> 6, c = t & 63;
  float s = gb[c];
  for (int ic = 0; ic < 64; ++ic) s += gw[c * 64 + ic] * pooled[b * 64 + ic];
  gate[t] = sigm_f(s);
}

__global__ void k_fcgate(const float* Fc2T, const float* gate, float* FcgT, _Float16* agfcat){
  int i = blockIdx.x * 256 + threadIdx.x;       // [b][n][c]
  int c = i & 63; int b = i >> 18;
  float v = Fc2T[i] * gate[(b << 6) + c];
  FcgT[i] = v;
  agfcat[(size_t)(i >> 6) * 128 + c] = (_Float16)v;
}

__global__ void k_fh(const float* aOutT, const float* maskT, float* FhT, _Float16* agfcat){
  int i = blockIdx.x * 256 + threadIdx.x;
  float v = aOutT[i] * maskT[i];
  FhT[i] = v;
  agfcat[(size_t)(i >> 6) * 128 + 64 + (i & 63)] = (_Float16)v;
}

__global__ void k_ff_ln(const float* gT, const float* FcgT, const float* FhT,
                        const float* lng, const float* lnb, _Float16* xn16){
  int p = blockIdx.x * 256 + threadIdx.x;       // B*HW pixels
  size_t base = (size_t)p << 6;
  float sum = 0.f, sq = 0.f;
  for (int c = 0; c < 64; ++c){
    float g = gT[base + c];
    float f = g * FcgT[base + c] + (1.f - g) * FhT[base + c];
    sum += f; sq += f * f;
  }
  float mu  = sum * (1.f / 64.f);
  float var = sq * (1.f / 64.f) - mu * mu;
  float inv = rsqrtf(var + 1e-5f);
  for (int c = 0; c < 64; ++c){
    float g = gT[base + c];
    float f = g * FcgT[base + c] + (1.f - g) * FhT[base + c];
    xn16[base + c] = (_Float16)((f - mu) * inv * lng[c] + lnb[c]);
  }
}

// mixed3/5/7 channel shuffle + depthwise k=3/5/7 + gelu -> refcat [HW][576]
__global__ void k_dw_mix(const _Float16* s3, const _Float16* s5, const _Float16* s7,
                         const float* r3w, const float* r3b,
                         const float* r5w, const float* r5b,
                         const float* r7w, const float* r7b,
                         _Float16* refcat){
  int i = blockIdx.x * 256 + threadIdx.x;       // B*HW*576
  if (i >= NB * HWSZ * 576) return;
  int ch = i % 576; int p = i / 576;
  int n = p & 4095; int b = p >> 12;
  int m = ch / 192; int c192 = ch - m * 192; int grp = c192 >> 6;
  int sel = (m + grp) % 3;
  const _Float16* src = (sel == 0) ? s3 : ((sel == 1) ? s5 : s7);
  const float* w  = (m == 0) ? r3w : ((m == 1) ? r5w : r7w);
  const float* bb = (m == 0) ? r3b : ((m == 1) ? r5b : r7b);
  int KS = 3 + 2 * m, pad = 1 + m;
  int y = n >> 6, x = n & 63;
  float s = bb[c192];
  for (int ky = 0; ky < KS; ++ky){
    int sy = y + ky - pad; if ((unsigned)sy >= 64u) continue;
    for (int kx = 0; kx < KS; ++kx){
      int sx = x + kx - pad; if ((unsigned)sx >= 64u) continue;
      s += (float)src[((size_t)((b << 12) + (sy << 6) + sx)) * 192 + c192]
           * w[c192 * KS * KS + ky * KS + kx];
    }
  }
  refcat[(size_t)p * 576 + ch] = (_Float16)gelu_f(s);
}

__global__ void k_final(const float* X, const float* FoutT, const float* rs, float* out){
  int i = blockIdx.x * 256 + threadIdx.x;       // NCHW
  int n = i & 4095; int c = (i >> 12) & 63; int b = i >> 18;
  out[i] = X[i] + rs[0] * FoutT[((((size_t)b << 12) + n) << 6) + c];
}

// ---------------------------------------------------------------------------
extern "C" void kernel_launch(void* const* d_in, const int* in_sizes, int n_in,
                              void* d_out, int out_size, void* d_ws, size_t ws_size,
                              hipStream_t stream) {
  (void)in_sizes; (void)n_in; (void)out_size; (void)ws_size;
  const float* X     = (const float*)d_in[0];
  const float* cwq_w = (const float*)d_in[1];  const float* cwq_b = (const float*)d_in[2];
  const float* cwk_w = (const float*)d_in[3];  const float* cwk_b = (const float*)d_in[4];
  const float* cwv_w = (const float*)d_in[5];  const float* cwv_b = (const float*)d_in[6];
  const float* cwr_w = (const float*)d_in[7];  const float* cwr_b = (const float*)d_in[8];
  const float* hsq_w = (const float*)d_in[9];  const float* hsq_b = (const float*)d_in[10];
  const float* hsk_w = (const float*)d_in[11]; const float* hsk_b = (const float*)d_in[12];
  const float* hsv_w = (const float*)d_in[13]; const float* hsv_b = (const float*)d_in[14];
  const float* cwg_w = (const float*)d_in[15]; const float* cwg_b = (const float*)d_in[16];
  const float* hsl_w = (const float*)d_in[17]; const float* hsl_b = (const float*)d_in[18];
  const float* hsc_w = (const float*)d_in[19]; const float* hsc_b = (const float*)d_in[20];
  const float* hsd_w = (const float*)d_in[21]; const float* hsd_b = (const float*)d_in[22];
  const float* agf_w = (const float*)d_in[23]; const float* agf_b = (const float*)d_in[24];
  const float* ln_g  = (const float*)d_in[25]; const float* ln_b  = (const float*)d_in[26];
  const float* e3_w  = (const float*)d_in[27]; const float* e3_b  = (const float*)d_in[28];
  const float* e5_w  = (const float*)d_in[29]; const float* e5_b  = (const float*)d_in[30];
  const float* e7_w  = (const float*)d_in[31]; const float* e7_b  = (const float*)d_in[32];
  const float* r3_w  = (const float*)d_in[33]; const float* r3_b  = (const float*)d_in[34];
  const float* r5_w  = (const float*)d_in[35]; const float* r5_b  = (const float*)d_in[36];
  const float* r7_w  = (const float*)d_in[37]; const float* r7_b  = (const float*)d_in[38];
  const float* fuse_w= (const float*)d_in[39]; const float* fuse_b= (const float*)d_in[40];
  const float* res_s = (const float*)d_in[41];
  float* outP = (float*)d_out;

  char* w8 = (char*)d_ws;
  size_t off = 0;
  auto alloc = [&](size_t bytes) -> char* {
    char* p = w8 + off;
    off = (off + bytes + 255) & ~(size_t)255;
    return p;
  };

  const size_t NHC = (size_t)NB * HWSZ * CD;          // 1M elems
  _Float16* Qc   = (_Float16*)alloc(NHC * 2);
  _Float16* Kc   = (_Float16*)alloc(NHC * 2);
  _Float16* VcT  = (_Float16*)alloc(NHC * 2);
  _Float16* qTt  = (_Float16*)alloc(NHC * 2);
  _Float16* kTt  = (_Float16*)alloc(NHC * 2);
  _Float16* vCc  = (_Float16*)alloc(NHC * 2);
  float*    Sc   = (float*)   alloc((size_t)NB * 64 * 64 * 4);
  _Float16* attn16 = (_Float16*)alloc((size_t)NB * 64 * 64 * 2);
  float*    FcT  = (float*)   alloc(NHC * 4);
  float*    Fc2T = (float*)   alloc(NHC * 4);
  float*    pooled = (float*) alloc(NB * 64 * 4);
  float*    gate   = (float*) alloc(NB * 64 * 4);
  float*    aOutT  = (float*) alloc(NHC * 4);
  _Float16* aT16   = (_Float16*)alloc(NHC * 2);
  _Float16* catCH  = (_Float16*)alloc((size_t)NB * HWSZ * 128 * 2);
  float*    maskT  = (float*) alloc(NHC * 4);
  float*    FcgT   = (float*) alloc(NHC * 4);
  float*    FhT    = (float*) alloc(NHC * 4);
  _Float16* agfcat = (_Float16*)alloc((size_t)NB * HWSZ * 128 * 2);
  float*    gT     = (float*) alloc(NHC * 4);
  _Float16* xn16   = (_Float16*)alloc(NHC * 2);
  _Float16* s3b    = (_Float16*)alloc((size_t)NB * HWSZ * HIDC * 2);
  _Float16* s5b    = (_Float16*)alloc((size_t)NB * HWSZ * HIDC * 2);
  _Float16* s7b    = (_Float16*)alloc((size_t)NB * HWSZ * HIDC * 2);
  _Float16* refcat = (_Float16*)alloc((size_t)NB * HWSZ * 576 * 2);
  float*    FoutT  = (float*) alloc(NHC * 4);
  _Float16* hslr = (_Float16*)alloc((size_t)64 * 64 * 2);
  _Float16* hscr = (_Float16*)alloc((size_t)64 * 576 * 2);
  _Float16* hsdr = (_Float16*)alloc((size_t)64 * 1152 * 2);
  _Float16* agfr = (_Float16*)alloc((size_t)64 * 128 * 2);
  _Float16* e3r  = (_Float16*)alloc((size_t)192 * 576 * 2);
  _Float16* e5r  = (_Float16*)alloc((size_t)192 * 1600 * 2);
  _Float16* e7r  = (_Float16*)alloc((size_t)192 * 3136 * 2);
  _Float16* fuser= (_Float16*)alloc((size_t)64 * 576 * 2);

  auto grids = [](int tot){ return (tot + 255) / 256; };

  // weight conversion / reorder
  k_reorder_w<<<grids(64*64),     256, 0, stream>>>(hsl_w,  hslr,  64, 64, 1);
  k_reorder_w<<<grids(64*64*9),   256, 0, stream>>>(hsc_w,  hscr,  64, 64, 3);
  k_reorder_w<<<grids(64*128*9),  256, 0, stream>>>(hsd_w,  hsdr,  64, 128, 3);
  k_reorder_w<<<grids(64*128),    256, 0, stream>>>(agf_w,  agfr,  64, 128, 1);
  k_reorder_w<<<grids(192*64*9),  256, 0, stream>>>(e3_w,   e3r,   192, 64, 3);
  k_reorder_w<<<grids(192*64*25), 256, 0, stream>>>(e5_w,   e5r,   192, 64, 5);
  k_reorder_w<<<grids(192*64*49), 256, 0, stream>>>(e7_w,   e7r,   192, 64, 7);
  k_reorder_w<<<grids(64*576),    256, 0, stream>>>(fuse_w, fuser, 64, 576, 1);

  // six depthwise QKV convs -> f16 operand layouts
  k_dw6<<<grids(NB*CD*HWSZ), 256, 0, stream>>>(X,
      cwq_w, cwq_b, cwk_w, cwk_b, cwv_w, cwv_b,
      hsq_w, hsq_b, hsk_w, hsk_b, hsv_w, hsv_b,
      Qc, Kc, VcT, qTt, kTt, vCc);

  const float scale = 0.125f;   // 1/sqrt(64)

  // channel attention
  k_chan_qk<<<dim3(16, NB), 32, 0, stream>>>(Qc, Kc, Sc, scale);
  k_chan_softmax<<<1, 256, 0, stream>>>(Sc, attn16);
  k_chan_av<<<dim3(256, 4, NB), 32, 0, stream>>>(VcT, attn16, FcT);

  // spatial attention (flash)
  k_spatial_attn<<<dim3(256, NB), 32, 0, stream>>>(qTt, kTt, vCc, aOutT, aT16, scale);

  // CWSAB tail: cwr depthwise, pooled sigmoid gate
  k_dw_nc<<<grids(NB*HWSZ*CD), 256, 0, stream>>>(FcT, cwr_w, cwr_b, Fc2T);
  k_pool<<<NB*64, 256, 0, stream>>>(Fc2T, pooled);
  k_gate<<<1, 256, 0, stream>>>(pooled, cwg_w, cwg_b, gate);
  k_fcgate<<<grids(NB*HWSZ*CD), 256, 0, stream>>>(Fc2T, gate, FcgT, agfcat);

  // HSAB tail: local(1x1) + ctx(3x3) -> mask(3x3 dil2, sigmoid) -> Fh
  k_conv1x1<<<dim3(128, 4, NB), 32, 0, stream>>>(aT16, hslr, hsl_b,
      nullptr, 0, 0, catCH, 128, 0, 64, 0);
  k_convK<3, 6><<<dim3(128, 4, NB), 32, 0, stream>>>(aT16, hscr, hsc_b,
      nullptr, 0, 0, catCH, 128, 64, 1, 1, 0);
  k_convK<3, 7><<<dim3(128, 4, NB), 32, 0, stream>>>(catCH, hsdr, hsd_b,
      maskT, 64, 0, nullptr, 0, 0, 2, 2, 1);
  k_fh<<<grids(NB*HWSZ*CD), 256, 0, stream>>>(aOutT, maskT, FhT, agfcat);

  // AGF fusion gate + LayerNorm
  k_conv1x1<<<dim3(128, 4, NB), 32, 0, stream>>>(agfcat, agfr, agf_b,
      gT, 64, 0, nullptr, 0, 0, 128, 1);
  k_ff_ln<<<grids(NB*HWSZ), 256, 0, stream>>>(gT, FcgT, FhT, ln_g, ln_b, xn16);

  // MLFFB expand convs (gelu) -> s3/s5/s7 f16
  k_convK<3, 6><<<dim3(128, 12, NB), 32, 0, stream>>>(xn16, e3r, e3_b,
      nullptr, 0, 0, s3b, 192, 0, 1, 1, 2);
  k_convK<5, 6><<<dim3(128, 12, NB), 32, 0, stream>>>(xn16, e5r, e5_b,
      nullptr, 0, 0, s5b, 192, 0, 2, 1, 2);
  k_convK<7, 6><<<dim3(128, 12, NB), 32, 0, stream>>>(xn16, e7r, e7_b,
      nullptr, 0, 0, s7b, 192, 0, 3, 1, 2);

  // channel-shuffled depthwise refinement + gelu -> refcat, then fuse 1x1
  k_dw_mix<<<grids(NB*HWSZ*576), 256, 0, stream>>>(s3b, s5b, s7b,
      r3_w, r3_b, r5_w, r5_b, r7_w, r7_b, refcat);
  k_conv1x1<<<dim3(128, 4, NB), 32, 0, stream>>>(refcat, fuser, fuse_b,
      FoutT, 64, 0, nullptr, 0, 0, 576, 0);

  // residual
  k_final<<<grids(NB*CD*HWSZ), 256, 0, stream>>>(X, FoutT, res_s, outP);
}
